// GCNEncoder_43705587204594
// MI455X (gfx1250) — matrix operand order repaired and checked by reading, not compiled, not run on hardware
//
#include <hip/hip_runtime.h>
#include <hip/hip_bf16.h>

// ---------------------------------------------------------------------------
// GCN encoder on MI455X (gfx1250, wave32).
//
// Pipeline is HBM/atomic bound (~2.7 GB traffic vs ~8 GFLOP), so GEMMs use
// v_wmma_f32_16x16x32_f16 (f16 inputs, f32 accum) with on-the-fly conversion:
// x is streamed from HBM exactly once; W1/W2 are pre-converted per block into
// LDS in WMMA B-fragment order so each B fragment is one contiguous 32B
// ds_load per lane. SpMM uses the native global_atomic_add_f32 path.
// ---------------------------------------------------------------------------

typedef __attribute__((ext_vector_type(16))) _Float16 v16h;
typedef __attribute__((ext_vector_type(8)))  float    v8f;

// ---------------------------------------------------------------------------
// GEMM1: h[N,128] = x[N,256] @ W1[128,256]^T + b1
// One wave per 16-row strip; 8 N-tiles x 8 K-steps = 64 WMMAs per strip.
// ---------------------------------------------------------------------------
__global__ __launch_bounds__(128) void gemm1_kernel(
    const float* __restrict__ x, const float* __restrict__ W1,
    const float* __restrict__ b1, float* __restrict__ h, int nStrips)
{
  // B in fragment order: ldsB[chunk = 2*(k/32) + (k/8)%2][n][k%8 + 8*((k/16)%2)], 16 chunks * 128 n * 16
  __shared__ alignas(32) _Float16 ldsB[16 * 128 * 16]; // 64 KB
  const int tid = threadIdx.x;
  for (int i = tid; i < 128 * 256; i += 128) {
    const int n = i >> 8, k = i & 255;
    ldsB[((((k >> 5) << 1) + ((k >> 3) & 1)) << 11) + (n << 4) + (k & 7) + (((k >> 4) & 1) << 3)] = (_Float16)W1[i];
  }
  __syncthreads();

  const int lane  = tid & 31;
  const int strip = blockIdx.x * 4 + (tid >> 5);
  if (strip >= nStrips) return;           // wave-uniform: EXEC stays all-1s

  const int m      = lane & 15;           // row within tile
  const int koff8  = (lane >> 4) << 3;    // A-frag: upper half-wave -> K+8
  const int koff16 = koff8 << 1;          // B-frag: upper half-wave -> K+16

  const float* xr = x + (size_t)(strip * 16 + m) * 256;
  v8f acc[8] = {};

  #pragma unroll
  for (int kb = 0; kb < 256; kb += 32) {
    // A fragment (16x32 f16, ISA layout): per-lane two contiguous 8-float
    // chunks of this row, converted f32->f16.
    const float4 c0 = *(const float4*)(xr + kb + koff8);
    const float4 c1 = *(const float4*)(xr + kb + koff8 + 4);
    const float4 c2 = *(const float4*)(xr + kb + 16 + koff8);
    const float4 c3 = *(const float4*)(xr + kb + 16 + koff8 + 4);
    v16h a;
    a[0]=(_Float16)c0.x; a[1]=(_Float16)c0.y; a[2]=(_Float16)c0.z; a[3]=(_Float16)c0.w;
    a[4]=(_Float16)c1.x; a[5]=(_Float16)c1.y; a[6]=(_Float16)c1.z; a[7]=(_Float16)c1.w;
    a[8]=(_Float16)c2.x; a[9]=(_Float16)c2.y; a[10]=(_Float16)c2.z; a[11]=(_Float16)c2.w;
    a[12]=(_Float16)c3.x; a[13]=(_Float16)c3.y; a[14]=(_Float16)c3.z; a[15]=(_Float16)c3.w;

    const int chunk = (kb + koff16) >> 4;
    const v16h* bp = (const v16h*)&ldsB[chunk << 11];
    #pragma unroll
    for (int nt = 0; nt < 8; ++nt) {
      const v16h b = bp[(nt << 4) + m];   // one 32B contiguous fragment
      acc[nt] = __builtin_amdgcn_wmma_f32_16x16x32_f16(
          false, a, false, b, (short)0, acc[nt], false, false);
    }
  }

  // C/D layout: VGPR r -> M = r (+8 for upper half-wave), N = lane&15
  #pragma unroll
  for (int nt = 0; nt < 8; ++nt) {
    const int   col  = (nt << 4) + m;
    const float bias = b1[col];
    #pragma unroll
    for (int r = 0; r < 8; ++r) {
      const int row = strip * 16 + r + koff8;
      h[(size_t)row * 128 + col] = acc[nt][r] + bias;
    }
  }
}

// ---------------------------------------------------------------------------
// GEMM2: z[N,64] = relu(agg1)[N,128] @ W2[64,128]^T + b2  (ReLU fused on load)
// ---------------------------------------------------------------------------
__global__ __launch_bounds__(128) void gemm2_kernel(
    const float* __restrict__ hin, const float* __restrict__ W2,
    const float* __restrict__ b2, float* __restrict__ z, int nStrips)
{
  __shared__ alignas(32) _Float16 ldsB[8 * 64 * 16];   // 16 KB
  const int tid = threadIdx.x;
  for (int i = tid; i < 64 * 128; i += 128) {
    const int n = i >> 7, k = i & 127;
    ldsB[((((k >> 5) << 1) + ((k >> 3) & 1)) << 10) + (n << 4) + (k & 7) + (((k >> 4) & 1) << 3)] = (_Float16)W2[i];
  }
  __syncthreads();

  const int lane  = tid & 31;
  const int strip = blockIdx.x * 4 + (tid >> 5);
  if (strip >= nStrips) return;

  const int m      = lane & 15;
  const int koff8  = (lane >> 4) << 3;
  const int koff16 = koff8 << 1;

  const float* xr = hin + (size_t)(strip * 16 + m) * 128;
  v8f acc[4] = {};

  #pragma unroll
  for (int kb = 0; kb < 128; kb += 32) {
    float4 c0 = *(const float4*)(xr + kb + koff8);
    float4 c1 = *(const float4*)(xr + kb + koff8 + 4);
    float4 c2 = *(const float4*)(xr + kb + 16 + koff8);
    float4 c3 = *(const float4*)(xr + kb + 16 + koff8 + 4);
    v16h a;   // ReLU fused into the f32->f16 conversion
    a[0]=(_Float16)fmaxf(c0.x,0.f); a[1]=(_Float16)fmaxf(c0.y,0.f);
    a[2]=(_Float16)fmaxf(c0.z,0.f); a[3]=(_Float16)fmaxf(c0.w,0.f);
    a[4]=(_Float16)fmaxf(c1.x,0.f); a[5]=(_Float16)fmaxf(c1.y,0.f);
    a[6]=(_Float16)fmaxf(c1.z,0.f); a[7]=(_Float16)fmaxf(c1.w,0.f);
    a[8]=(_Float16)fmaxf(c2.x,0.f); a[9]=(_Float16)fmaxf(c2.y,0.f);
    a[10]=(_Float16)fmaxf(c2.z,0.f); a[11]=(_Float16)fmaxf(c2.w,0.f);
    a[12]=(_Float16)fmaxf(c3.x,0.f); a[13]=(_Float16)fmaxf(c3.y,0.f);
    a[14]=(_Float16)fmaxf(c3.z,0.f); a[15]=(_Float16)fmaxf(c3.w,0.f);

    const int chunk = (kb + koff16) >> 4;
    const v16h* bp = (const v16h*)&ldsB[chunk << 10];
    #pragma unroll
    for (int nt = 0; nt < 4; ++nt) {
      const v16h b = bp[(nt << 4) + m];
      acc[nt] = __builtin_amdgcn_wmma_f32_16x16x32_f16(
          false, a, false, b, (short)0, acc[nt], false, false);
    }
  }

  #pragma unroll
  for (int nt = 0; nt < 4; ++nt) {
    const int   col  = (nt << 4) + m;
    const float bias = b2[col];
    #pragma unroll
    for (int r = 0; r < 8; ++r) {
      const int row = strip * 16 + r + koff8;
      z[(size_t)row * 64 + col] = acc[nt][r] + bias;
    }
  }
}

// ---------------------------------------------------------------------------
// SpMM scatter: out[dst] += w * h[src].  One wave32 per edge; each lane owns
// C/32 channels (float4 for C=128, float2 for C=64) + global_atomic_add_f32.
// ---------------------------------------------------------------------------
template <int C>
__global__ __launch_bounds__(256) void spmm_kernel(
    const int* __restrict__ dst, const int* __restrict__ src,
    const float* __restrict__ ew, const float* __restrict__ hin,
    float* __restrict__ out, int E)
{
  const int e    = blockIdx.x * 8 + (threadIdx.x >> 5);
  const int lane = threadIdx.x & 31;
  if (e >= E) return;
  const int   d = dst[e];
  const int   s = src[e];
  const float w = ew[e];
  constexpr int PL = C / 32;  // floats per lane
  const float* hp = hin + (size_t)s * C + lane * PL;
  float*       op = out + (size_t)d * C + lane * PL;
  if constexpr (PL == 4) {
    const float4 v = *(const float4*)hp;
    unsafeAtomicAdd(op + 0, w * v.x);
    unsafeAtomicAdd(op + 1, w * v.y);
    unsafeAtomicAdd(op + 2, w * v.z);
    unsafeAtomicAdd(op + 3, w * v.w);
  } else {
    const float2 v = *(const float2*)hp;
    unsafeAtomicAdd(op + 0, w * v.x);
    unsafeAtomicAdd(op + 1, w * v.y);
  }
}

// ---------------------------------------------------------------------------
// Row L2-normalize [N,64] in place. One wave per node, wave32 shfl reduction.
// ---------------------------------------------------------------------------
__global__ __launch_bounds__(256) void normalize_kernel(float* __restrict__ z, int N)
{
  const int n    = blockIdx.x * 8 + (threadIdx.x >> 5);
  const int lane = threadIdx.x & 31;
  if (n >= N) return;
  float* p = z + (size_t)n * 64 + lane * 2;
  const float2 v = *(const float2*)p;
  float ss = v.x * v.x + v.y * v.y;
  #pragma unroll
  for (int off = 16; off; off >>= 1) ss += __shfl_xor(ss, off, 32);
  const float sc = 1.0f / fmaxf(sqrtf(ss), 1e-12f);
  float2 o; o.x = v.x * sc; o.y = v.y * sc;
  *(float2*)p = o;
}

// ---------------------------------------------------------------------------
extern "C" void kernel_launch(void* const* d_in, const int* in_sizes, int n_in,
                              void* d_out, int out_size, void* d_ws, size_t ws_size,
                              hipStream_t stream) {
  const float* x  = (const float*)d_in[0];
  const int*   ei = (const int*)  d_in[1];
  const float* ew = (const float*)d_in[2];
  const float* W1 = (const float*)d_in[3];
  const float* b1 = (const float*)d_in[4];
  const float* W2 = (const float*)d_in[5];
  const float* b2 = (const float*)d_in[6];

  const int N = in_sizes[0] / 256;   // 100000 (multiple of 16)
  const int E = in_sizes[2];         // 1600000
  const int* dstIdx = ei;            // edge_index[0]
  const int* srcIdx = ei + E;        // edge_index[1]
  float* zout = (float*)d_out;

  // Workspace: h [N,128] | agg1 [N,128];  z [N,64] reuses h's region.
  char*  ws   = (char*)d_ws;
  float* h    = (float*)ws;
  float* agg1 = (float*)(ws + (size_t)N * 128 * sizeof(float));
  float* z    = h;

  // Zero the atomic-accumulation targets every call (graph-capturable).
  hipMemsetAsync(agg1, 0, (size_t)N * 128 * sizeof(float), stream);
  hipMemsetAsync(zout, 0, (size_t)N * 64  * sizeof(float), stream);

  const int nStrips = (N + 15) / 16;       // 6250
  const int gBlocks = (nStrips + 3) / 4;   // 4 waves (strips) per block
  const int eBlocks = (E + 7) / 8;         // 8 edges (waves) per block
  const int nBlocks = (N + 7) / 8;         // 8 nodes (waves) per block

  gemm1_kernel<<<gBlocks, 128, 0, stream>>>(x, W1, b1, h, nStrips);
  spmm_kernel<128><<<eBlocks, 256, 0, stream>>>(dstIdx, srcIdx, ew, h, agg1, E);
  gemm2_kernel<<<gBlocks, 128, 0, stream>>>(agg1, W2, b2, z, nStrips);
  spmm_kernel<64><<<eBlocks, 256, 0, stream>>>(dstIdx, srcIdx, ew, z, zout, E);
  normalize_kernel<<<nBlocks, 256, 0, stream>>>(zout, N);
}
